// AttentionCNN_29755533426756
// MI455X (gfx1250) — compile-verified, matrix operand order checked by reading, not run on hardware
//
#include <hip/hip_runtime.h>
#include <hip/hip_bf16.h>
#include <math.h>

// ---------------- CDNA5 WMMA types ----------------
typedef __attribute__((ext_vector_type(16))) __bf16 v16bf;
typedef __attribute__((ext_vector_type(8)))  float  v8f;

union ABFrag {                 // one bf16 16x32 (A) or 32x16 (B) fragment per wave
    v16bf v;
    unsigned short u[16];
    uint4 q[2];
};
union AccFrag {                // 16x16 f32 C/D fragment
    v8f v;
    float f[8];
};

__device__ __forceinline__ unsigned short f32_to_bf16_raw(float f) {
    union { __bf16 b; unsigned short u; } cv;
    cv.b = (__bf16)f;
    return cv.u;
}

// ---------------- small utility kernels ----------------
__global__ void zero_f32_kernel(float* p, int n) {
    int i = blockIdx.x * 256 + threadIdx.x;
    if (i < n) p[i] = 0.f;
}

// x: (4,12,32,32) f32 NCHW  ->  actA: (4,32,32,32) bf16 NHWC, channels padded 12->32 with zeros
__global__ void repack_input_kernel(const float* __restrict__ x, unsigned short* __restrict__ actA) {
    int idx = blockIdx.x * 256 + threadIdx.x;      // 4*32*32*32 = 131072
    int c  = idx & 31;
    int sp = (idx >> 5) & 1023;                    // h*32+w
    int n  = idx >> 15;
    float v = (c < 12) ? x[((size_t)n * 12 + c) * 1024 + sp] : 0.f;
    actA[idx] = f32_to_bf16_raw(v);
}

// Repack OIHW f32 weights into per-lane WMMA B-fragment order (bf16):
// frag[((tap*KB+kb)*NB+nb)*32 + lane][16]
__global__ void repack_weights_kernel(const float* __restrict__ w, unsigned short* __restrict__ frag,
                                      int I, int O, int KB, int NB) {
    int idx = blockIdx.x * 256 + threadIdx.x;
    int total = 9 * KB * NB * 512;
    if (idx >= total) return;
    int e    = idx & 15;
    int lane = (idx >> 4) & 31;
    int r2   = idx >> 9;                 // (tap*KB+kb)*NB+nb
    int nb   = r2 % NB;
    int rest = r2 / NB;                  // tap*KB+kb
    int kb   = rest % KB;
    int tap  = rest / KB;
    // B layout mirror of A: lane<16 holds K chunks {0..7,16..23}; lane>=16 holds {8..15,24..31}
    int chunk0 = (lane < 16) ? 0 : 8;
    int kl  = (e < 8) ? (chunk0 + e) : (chunk0 + 16 + (e - 8));
    int cin = kb * 32 + kl;
    int o   = nb * 16 + (lane & 15);
    int dy  = tap / 3, dx = tap % 3;
    float v = 0.f;
    if (cin < I && o < O) v = w[(((size_t)o * I + cin) * 3 + dy) * 3 + dx];
    frag[idx] = f32_to_bf16_raw(v);
}

// ---------------- implicit-GEMM 3x3 conv via bf16 WMMA ----------------
// Workgroup: 8 waves; computes 4 output rows x 32 cols (M=128) x NB*16 out-channels for image n.
// Input NHWC bf16 (C = CP), staged rows h0-1 .. h0+4 into LDS.
template <int CP, int KB, int NB, bool TSTORE>
__global__ void conv3x3_wmma_kernel(const unsigned short* __restrict__ actin,
                                    const unsigned short* __restrict__ wfrag,
                                    const float* __restrict__ bias,
                                    float* __restrict__ yout,
                                    float* __restrict__ stats,   // [sum(64), sumsq(64)] or nullptr
                                    int cout) {
    constexpr int ROWE = 32 * CP;                 // ushorts per image row
    __shared__ unsigned short lin[6 * ROWE];

    const int n  = blockIdx.x >> 3;
    const int rb = blockIdx.x & 7;
    const int h0 = rb * 4;

    // stage 6 input rows (zero for out-of-image rows)
    for (int idx = threadIdx.x; idx < 6 * ROWE / 8; idx += 256) {
        int r   = idx / (ROWE / 8);
        int off = (idx % (ROWE / 8)) * 8;
        int hh  = h0 - 1 + r;
        uint4 v = uint4{0u, 0u, 0u, 0u};
        if (hh >= 0 && hh < 32)
            v = *(const uint4*)(actin + (size_t)(n * 32 + hh) * ROWE + off);
        *(uint4*)(lin + r * ROWE + off) = v;
    }
    __syncthreads();

    const int wave = threadIdx.x >> 5;
    const int lane = threadIdx.x & 31;
    const int mrow = lane & 15;
    const int pos  = wave * 16 + mrow;            // local spatial position of this lane's A row
    const int prow = pos >> 5;
    const int pcol = pos & 31;
    const int ncol = lane & 15;
    const int k0   = (lane < 16) ? 0 : 8;         // A chunk base per wave32 bf16 layout

    AccFrag acc[NB];
#pragma unroll
    for (int nb = 0; nb < NB; ++nb) {
        int ch  = nb * 16 + ncol;
        float b = (ch < cout) ? bias[ch] : 0.f;
#pragma unroll
        for (int e = 0; e < 8; ++e) acc[nb].f[e] = b;
    }

    for (int tap = 0; tap < 9; ++tap) {
        const int dy = tap / 3, dx = tap % 3;
        const int r = prow + dy;                  // LDS row (maps to h0+prow+dy-1)
        const int c = pcol + dx - 1;
        const bool valid = (c >= 0) && (c < 32);
#pragma unroll
        for (int kb = 0; kb < KB; ++kb) {
            ABFrag a;
            if (valid) {
                const unsigned short* p = lin + (r * 32 + c) * CP + kb * 32 + k0;
#pragma unroll
                for (int e = 0; e < 8; ++e) { a.u[e] = p[e]; a.u[e + 8] = p[e + 16]; }
            } else {
                a.q[0] = uint4{0u, 0u, 0u, 0u};
                a.q[1] = uint4{0u, 0u, 0u, 0u};
            }
#pragma unroll
            for (int nb = 0; nb < NB; ++nb) {
                ABFrag b;
                const uint4* q = (const uint4*)(wfrag + ((((size_t)tap * KB + kb) * NB + nb) * 32 + lane) * 16);
                b.q[0] = q[0];
                b.q[1] = q[1];
                acc[nb].v = __builtin_amdgcn_wmma_f32_16x16x32_bf16(
                    false, a.v, false, b.v, (short)0, acc[nb].v, false, false);
            }
        }
    }

    // epilogue: store + (optional) per-channel batch-stat atomics
#pragma unroll
    for (int nb = 0; nb < NB; ++nb) {
        const int ch = nb * 16 + ncol;
        if (TSTORE) {
            if (ch < cout) {
#pragma unroll
                for (int rr = 0; rr < 8; ++rr) {
                    int M = rr + ((lane >> 4) << 3);
                    int p = wave * 16 + M;                       // local spatial
                    yout[((size_t)n * cout + ch) * 1024 + h0 * 32 + p] = acc[nb].f[rr];
                }
            }
        } else {
            float s = 0.f, qq = 0.f;
#pragma unroll
            for (int rr = 0; rr < 8; ++rr) {
                int M = rr + ((lane >> 4) << 3);
                int p = wave * 16 + M;
                int row = h0 + (p >> 5);
                int col = p & 31;
                float v = acc[nb].f[rr];
                yout[((size_t)(n * 32 + row) * 32 + col) * 64 + ch] = v;
                s += v;
                qq += v * v;
            }
            if (stats) {
                atomicAdd(stats + ch, s);
                atomicAdd(stats + 64 + ch, qq);
            }
        }
    }
}

// ---------------- BN (batch stats) + ReLU + f32->bf16 ----------------
__global__ void bnrelu_kernel(const float* __restrict__ y, const float* __restrict__ stats,
                              const float* __restrict__ gamma, const float* __restrict__ beta,
                              unsigned short* __restrict__ actout) {
    int idx = blockIdx.x * 256 + threadIdx.x;      // 4*32*32*64 = 262144 (NHWC)
    int c = idx & 63;
    float mean = stats[c] * (1.f / 4096.f);
    float var  = stats[64 + c] * (1.f / 4096.f) - mean * mean;
    float sc   = gamma[c] * rsqrtf(var + 1e-5f);
    float v    = (y[idx] - mean) * sc + beta[c];
    v = fmaxf(v, 0.f);
    actout[idx] = f32_to_bf16_raw(v);
}

// ---------------- Hadamard attention: scores=w*x, softmax, out=att@x ----------------
// One wave per softmax row (1024 elems -> 32/lane). Grid ordered (t, iblk, n) so the 4
// n-replicas of each att_w row hit L1/L2 back-to-back; att output (201 MB) streamed NT.
__global__ void attention_kernel(const float* __restrict__ attw, const float* __restrict__ xf,
                                 float* __restrict__ out, float* __restrict__ att) {
    __shared__ float xs[1024];
    const int bid  = blockIdx.x;                   // 12*128*4 = 6144
    const int n    = bid & 3;
    const int rest = bid >> 2;
    const int iblk = rest & 127;
    const int t    = rest >> 7;
    const int nt   = n * 12 + t;

    const float* xrow = xf + (size_t)nt * 1024;
    for (int j = threadIdx.x; j < 1024; j += 256) xs[j] = xrow[j];
    __syncthreads();

    const int wave = threadIdx.x >> 5;
    const int lane = threadIdx.x & 31;
    const int i    = iblk * 8 + wave;
    const float* wrow = attw + ((size_t)t * 1024 + i) * 1024;

    float s[32];
    float m = -INFINITY;
#pragma unroll
    for (int k = 0; k < 32; ++k) {
        int j = k * 32 + lane;
        if ((k & 7) == 0) __builtin_prefetch(wrow + j + 512, 0, 0);  // global_prefetch_b8
        float v = wrow[j] * xs[j];
        s[k] = v;
        m = fmaxf(m, v);
    }
#pragma unroll
    for (int o = 16; o > 0; o >>= 1) m = fmaxf(m, __shfl_xor(m, o, 32));

    float sum = 0.f;
#pragma unroll
    for (int k = 0; k < 32; ++k) {
        float e = __expf(s[k] - m);
        s[k] = e;
        sum += e;
    }
#pragma unroll
    for (int o = 16; o > 0; o >>= 1) sum += __shfl_xor(sum, o, 32);
    const float inv = 1.f / sum;

    float oacc = 0.f;
    float* arow = att + ((size_t)nt * 1024 + i) * 1024;
#pragma unroll
    for (int k = 0; k < 32; ++k) {
        int j = k * 32 + lane;
        float a = s[k] * inv;
        __builtin_nontemporal_store(a, arow + j);   // don't pollute L2 (att_w is the reusable set)
        oacc += a * xs[j];
    }
#pragma unroll
    for (int o = 16; o > 0; o >>= 1) oacc += __shfl_xor(oacc, o, 32);
    if (lane == 0) out[(size_t)nt * 1024 + i] = oacc;
}

// ---------------- host-side orchestration ----------------
extern "C" void kernel_launch(void* const* d_in, const int* in_sizes, int n_in,
                              void* d_out, int out_size, void* d_ws, size_t ws_size,
                              hipStream_t stream) {
    const float* x       = (const float*)d_in[0];   // (4,12,32,32)
    const float* conv0_w = (const float*)d_in[1];   // (64,12,3,3)
    const float* conv_w  = (const float*)d_in[2];   // (4,64,64,3,3)
    const float* conv_b  = (const float*)d_in[3];   // (5,64)
    const float* bn_g    = (const float*)d_in[4];   // (5,64)
    const float* bn_b    = (const float*)d_in[5];   // (5,64)
    const float* reg_w   = (const float*)d_in[6];   // (12,64,3,3)
    const float* reg_b   = (const float*)d_in[7];   // (12,)
    const float* att_w   = (const float*)d_in[8];   // (12,1024,1024)

    float* out_main = (float*)d_out;                // (4,12,32,32)
    float* out_att  = out_main + 4 * 12 * 1024;     // (4,12,1024,1024)

    // workspace layout (bytes)
    char* w = (char*)d_ws;
    float*          stats = (float*)w;                               // 5*128 f32
    unsigned short* actA  = (unsigned short*)(w + 4096);             // 4*32*32*32 bf16  = 262144 B
    unsigned short* act   = (unsigned short*)(w + 4096 + 262144);    // 4*32*32*64 bf16  = 524288 B
    float*          y     = (float*)(w + 4096 + 262144 + 524288);    // 4*32*32*64 f32   = 1048576 B
    float*          yreg  = (float*)((char*)y + 1048576);            // 4*12*1024 f32    = 196608 B
    unsigned short* wf0   = (unsigned short*)((char*)yreg + 196608); // 18432 bf16       = 36864 B
    unsigned short* wfl   = (unsigned short*)((char*)wf0 + 36864);   // 4 * 36864 bf16   = 294912 B
    unsigned short* wfr   = (unsigned short*)((char*)wfl + 294912);  // 9216 bf16        = 18432 B

    // 1) zero BN-stat accumulators (harness does not re-zero ws between replays)
    zero_f32_kernel<<<3, 256, 0, stream>>>(stats, 640);

    // 2) pack input + all weights into WMMA fragment layouts
    repack_input_kernel<<<512, 256, 0, stream>>>(x, actA);
    repack_weights_kernel<<<72, 256, 0, stream>>>(conv0_w, wf0, 12, 64, 1, 4);
    for (int l = 0; l < 4; ++l)
        repack_weights_kernel<<<144, 256, 0, stream>>>(conv_w + (size_t)l * 64 * 64 * 9,
                                                       wfl + (size_t)l * 36864, 64, 64, 2, 4);
    repack_weights_kernel<<<36, 256, 0, stream>>>(reg_w, wfr, 64, 12, 2, 1);

    // 3) conv block: 5 x (WMMA conv + stats -> BN+ReLU -> bf16 activations)
    conv3x3_wmma_kernel<32, 1, 4, false><<<32, 256, 0, stream>>>(actA, wf0, conv_b, y, stats, 64);
    bnrelu_kernel<<<1024, 256, 0, stream>>>(y, stats, bn_g, bn_b, act);
    for (int l = 1; l < 5; ++l) {
        conv3x3_wmma_kernel<64, 2, 4, false><<<32, 256, 0, stream>>>(
            act, wfl + (size_t)(l - 1) * 36864, conv_b + l * 64, y, stats + l * 128, 64);
        bnrelu_kernel<<<1024, 256, 0, stream>>>(y, stats + l * 128, bn_g + l * 64, bn_b + l * 64, act);
    }

    // 4) regressor conv 64 -> 12, stored transposed as xf[n][t][hw]
    conv3x3_wmma_kernel<64, 2, 1, true><<<32, 256, 0, stream>>>(act, wfr, reg_b, yreg, nullptr, 12);

    // 5) attention (memory-bound stage: 201 MB NT write of att, att_w L2-resident)
    attention_kernel<<<6144, 256, 0, stream>>>(att_w, yreg, out_main, out_att);
}
